// P_RNN_86509231276755
// MI455X (gfx1250) — matrix operand order repaired and checked by reading, not compiled, not run on hardware
//
#include <hip/hip_runtime.h>
#include <hip/hip_bf16.h>

typedef __attribute__((ext_vector_type(16))) __bf16 v16bf;
typedef __attribute__((ext_vector_type(8)))  __bf16 v8bf;
typedef __attribute__((ext_vector_type(8)))  float  v8f;

#define T_LEN  32
#define BATCH  256
#define HID    512
#define XSV    512
#define RDEPTH 4
#define KW     512            // every GEMM in this net has K = 512

// LDS panel geometry: 16 rows x 512 bf16, each row padded by 8 bf16 (16B) so the
// row stride is 260 dwords == 4 (mod 64): each lane-half of a ds_load_b128 covers
// all 64 LDS banks exactly once -> structurally conflict-free.
#define ROW_BYTES   1040      // (512 + 8) * 2
#define PANEL_BYTES (16 * ROW_BYTES)
#define GRU_LDS     (6 * PANEL_BYTES)   // 99,840 B (under 320KB WGP LDS)
#define RO_LDS      (1 * PANEL_BYTES)   // 16,640 B

#define WMMA_BF16(a, b, c) \
  __builtin_amdgcn_wmma_f32_16x16x32_bf16(false, (a), false, (b), (short)0, (c), false, false)

__device__ __forceinline__ float prnn_sigmoid(float x) { return 1.0f / (1.0f + __expf(-x)); }

// ---- CDNA5 async copy: global -> LDS, tracked by ASYNCcnt ------------------
__device__ __forceinline__ void prnn_async_b128(const void* gsrc, void* ldst) {
  unsigned lds = (unsigned)(size_t)ldst;                 // flat aperture: addr[31:0] = LDS offset
  unsigned long long g = (unsigned long long)(size_t)gsrc;
  asm volatile("global_load_async_to_lds_b128 %0, %1, off" :: "v"(lds), "v"(g) : "memory");
}
__device__ __forceinline__ void prnn_wait_async0() {
  asm volatile("s_wait_asynccnt 0x0" ::: "memory");
}

// A fragment (16x32 bf16, M x K): lane half 0 -> K {k0..k0+7, k0+16..k0+23},
// lane half 1 -> K {k0+8..k0+15, k0+24..k0+31}; row M = m0 + (lane & 15).
__device__ __forceinline__ v16bf prnn_load_a(const __hip_bfloat16* base, int m0, int k0, int lane) {
  const int half = lane >> 4;
  const int row  = lane & 15;
  const __bf16* p = reinterpret_cast<const __bf16*>(base) + (size_t)(m0 + row) * KW + k0 + half * 8;
  v8bf lo = *reinterpret_cast<const v8bf*>(p);
  v8bf hi = *reinterpret_cast<const v8bf*>(p + 16);
  return __builtin_shufflevector(lo, hi, 0, 1, 2, 3, 4, 5, 6, 7, 8, 9, 10, 11, 12, 13, 14, 15);
}

// B fragment (32x16 bf16, K x N) sourced from an LDS-staged (N,K) panel:
// lane holds one column of B == one padded row of the panel.
__device__ __forceinline__ v16bf prnn_lds_b(const char* panel, int k0, int lane) {
  const int half = lane >> 4;
  const int col  = lane & 15;
  const __bf16* p = reinterpret_cast<const __bf16*>(panel + col * ROW_BYTES) + k0 + half * 8;
  v8bf lo = *reinterpret_cast<const v8bf*>(p);
  v8bf hi = *reinterpret_cast<const v8bf*>(p + 16);
  return __builtin_shufflevector(lo, hi, 0, 1, 2, 3, 4, 5, 6, 7, 8, 9, 10, 11, 12, 13, 14, 15);
}

// -------- fused GRU cell ----------------------------------------------------
// block = 128 (4 waves), grid = (BATCH/128, HID/16). Workgroup stages the six
// 16-row weight panels (Wih/Whh x r,z,n) into LDS via async copies; each wave
// computes two 16x16 h' tiles (12 fp32 WMMA accumulators).
__global__ __launch_bounds__(128)
void prnn_gru_cell(const __hip_bfloat16* __restrict__ z,
                   const __hip_bfloat16* __restrict__ hin,
                   const __hip_bfloat16* __restrict__ Wih,   // (3H, K) bf16
                   const __hip_bfloat16* __restrict__ Whh,   // (3H, K) bf16
                   const float* __restrict__ bih,
                   const float* __restrict__ bhh,
                   __hip_bfloat16* __restrict__ hout,
                   float* __restrict__ hout_f32)             // nullable fp32 shadow
{
  extern __shared__ char smem[];
  const int tid  = threadIdx.x;
  const int lane = tid & 31;
  const int wid  = tid >> 5;
  const int n0 = blockIdx.y * 16;                 // gate-column tile
  const int m0 = blockIdx.x * 128 + wid * 16;     // first of two batch tiles (+64)

  // ---- stage 6 weight panels: 6 * 16 rows * 64 16B-chunks = 6144 chunks
  for (int c = tid; c < 6 * 16 * 64; c += 128) {
    const int panel = c >> 10;          // / (16*64)
    const int rem   = c & 1023;
    const int r     = rem >> 6;
    const int ck    = rem & 63;
    const __hip_bfloat16* src =
        (panel < 3 ? Wih : Whh) + (size_t)(n0 + (panel % 3) * HID + r) * KW + ck * 8;
    prnn_async_b128(src, smem + panel * PANEL_BYTES + r * ROW_BYTES + ck * 16);
  }
  prnn_wait_async0();
  __syncthreads();

  v8f accA[2][3] = {};   // z @ Wih^T   (two m-tiles x r,z,n)
  v8f accB[2][3] = {};   // h @ Whh^T

  for (int k0 = 0; k0 < KW; k0 += 32) {
    v16bf aZ[2], aH[2];
    aZ[0] = prnn_load_a(z,   m0,      k0, lane);
    aZ[1] = prnn_load_a(z,   m0 + 64, k0, lane);
    aH[0] = prnn_load_a(hin, m0,      k0, lane);
    aH[1] = prnn_load_a(hin, m0 + 64, k0, lane);
    if (k0 + 64 <= KW - 32) {  // keep the activation stream warm
      __builtin_prefetch((const void*)(z   + (size_t)m0 * KW + k0 + 64), 0, 3);
      __builtin_prefetch((const void*)(hin + (size_t)m0 * KW + k0 + 64), 0, 3);
    }
#pragma unroll
    for (int g = 0; g < 3; ++g) {
      v16bf bi = prnn_lds_b(smem + g * PANEL_BYTES,       k0, lane);
      v16bf bh = prnn_lds_b(smem + (3 + g) * PANEL_BYTES, k0, lane);
#pragma unroll
      for (int mt = 0; mt < 2; ++mt) {
        accA[mt][g] = WMMA_BF16(aZ[mt], bi, accA[mt][g]);
        accB[mt][g] = WMMA_BF16(aH[mt], bh, accB[mt][g]);
      }
    }
  }

  // C/D layout: vgpr r, lanes 0-15 -> (M=r, N=lane); lanes 16-31 -> (M=r+8, N=lane-16)
  const int half = lane >> 4;
  const int col  = lane & 15;
  const int j = n0 + col;
  const float bir = bih[j],            bhr = bhh[j];
  const float biz = bih[HID + j],      bhz = bhh[HID + j];
  const float bin_ = bih[2 * HID + j], bhn = bhh[2 * HID + j];
#pragma unroll
  for (int mt = 0; mt < 2; ++mt) {
#pragma unroll
    for (int r = 0; r < 8; ++r) {
      const int m = m0 + mt * 64 + r + half * 8;
      const float rg = prnn_sigmoid(accA[mt][0][r] + bir + accB[mt][0][r] + bhr);
      const float zg = prnn_sigmoid(accA[mt][1][r] + biz + accB[mt][1][r] + bhz);
      const float ng = tanhf(accA[mt][2][r] + bin_ + rg * (accB[mt][2][r] + bhn));
      const float hp = __bfloat162float(hin[(size_t)m * HID + j]);
      const float hv = (1.0f - zg) * ng + zg * hp;
      hout[(size_t)m * HID + j] = __float2bfloat16(hv);
      if (hout_f32) hout_f32[(size_t)m * HID + j] = hv;
    }
  }
}

// -------- readout GEMM: out = act(A @ W^T + b) ------------------------------
// block = 128 (4 waves), grid = (BATCH/128, 512/16); one staged weight panel,
// two m-tiles per wave; dual bf16 / strided-fp32 stores.
__global__ __launch_bounds__(128)
void prnn_gemm_bias(const __hip_bfloat16* __restrict__ A,
                    const __hip_bfloat16* __restrict__ W,   // (512, 512) bf16
                    const float* __restrict__ bias,
                    __hip_bfloat16* __restrict__ out_bf,    // nullable, stride 512
                    float* __restrict__ out_f32,            // nullable
                    int f32_stride, int relu)
{
  extern __shared__ char smem[];
  const int tid  = threadIdx.x;
  const int lane = tid & 31;
  const int wid  = tid >> 5;
  const int n0 = blockIdx.y * 16;
  const int m0 = blockIdx.x * 128 + wid * 16;

  for (int c = tid; c < 16 * 64; c += 128) {   // 1024 16B chunks
    const int r  = c >> 6;
    const int ck = c & 63;
    prnn_async_b128(W + (size_t)(n0 + r) * KW + ck * 8, smem + r * ROW_BYTES + ck * 16);
  }
  prnn_wait_async0();
  __syncthreads();

  v8f acc[2] = {};
  for (int k0 = 0; k0 < KW; k0 += 32) {
    v16bf a0 = prnn_load_a(A, m0,      k0, lane);
    v16bf a1 = prnn_load_a(A, m0 + 64, k0, lane);
    v16bf b  = prnn_lds_b(smem, k0, lane);
    acc[0] = WMMA_BF16(a0, b, acc[0]);
    acc[1] = WMMA_BF16(a1, b, acc[1]);
  }

  const int half = lane >> 4;
  const int col  = lane & 15;
  const int j = n0 + col;
  const float bv = bias[j];
#pragma unroll
  for (int mt = 0; mt < 2; ++mt) {
#pragma unroll
    for (int r = 0; r < 8; ++r) {
      const int m = m0 + mt * 64 + r + half * 8;
      float v = acc[mt][r] + bv;
      if (relu) v = v > 0.0f ? v : 0.0f;
      if (out_bf)  out_bf[(size_t)m * 512 + j] = __float2bfloat16(v);
      if (out_f32) out_f32[(size_t)m * f32_stride + j] = v;
    }
  }
}

// -------- utility kernels ---------------------------------------------------
__global__ void prnn_f32_to_bf16(const float* __restrict__ in, __hip_bfloat16* __restrict__ out, int n) {
  int i = blockIdx.x * blockDim.x + threadIdx.x;
  if (i < n) out[i] = __float2bfloat16(in[i]);
}
__global__ void prnn_zero_bf16(__hip_bfloat16* __restrict__ p, int n) {
  int i = blockIdx.x * blockDim.x + threadIdx.x;
  if (i < n) p[i] = __float2bfloat16(0.0f);
}
__global__ void prnn_copy_f32(const float* __restrict__ in, float* __restrict__ out, int n) {
  int i = blockIdx.x * blockDim.x + threadIdx.x;
  if (i < n) out[i] = in[i];
}

// ---------------------------------------------------------------------------
extern "C" void kernel_launch(void* const* d_in, const int* in_sizes, int n_in,
                              void* d_out, int out_size, void* d_ws, size_t ws_size,
                              hipStream_t stream) {
  const float* x = (const float*)d_in[0];
  const float* Wih[3] = {(const float*)d_in[1], (const float*)d_in[5], (const float*)d_in[9]};
  const float* Whh[3] = {(const float*)d_in[2], (const float*)d_in[6], (const float*)d_in[10]};
  const float* bih[3] = {(const float*)d_in[3], (const float*)d_in[7], (const float*)d_in[11]};
  const float* bhh[3] = {(const float*)d_in[4], (const float*)d_in[8], (const float*)d_in[12]};
  const float* W1 = (const float*)d_in[13]; const float* b1 = (const float*)d_in[14];
  const float* W2 = (const float*)d_in[15]; const float* b2 = (const float*)d_in[16];
  const float* W3 = (const float*)d_in[17]; const float* b3 = (const float*)d_in[18];
  float* out = (float*)d_out;

  // ---- carve workspace (bf16 weights resident in 192MB L2: ~11.5 MB total)
  char* p = (char*)d_ws;
  auto carve = [&](size_t bytes) -> char* {
    char* r = p; p += (bytes + 255) & ~(size_t)255; return r;
  };
  __hip_bfloat16 *wihb[3], *whhb[3];
  for (int l = 0; l < 3; ++l) wihb[l] = (__hip_bfloat16*)carve((size_t)3 * HID * KW * 2);
  for (int l = 0; l < 3; ++l) whhb[l] = (__hip_bfloat16*)carve((size_t)3 * HID * KW * 2);
  __hip_bfloat16* w1b = (__hip_bfloat16*)carve((size_t)HID * KW * 2);
  __hip_bfloat16* w2b = (__hip_bfloat16*)carve((size_t)HID * KW * 2);
  __hip_bfloat16* w3b = (__hip_bfloat16*)carve((size_t)XSV * KW * 2);
  __hip_bfloat16* xb  = (__hip_bfloat16*)carve((size_t)T_LEN * BATCH * XSV * 2);
  __hip_bfloat16* hbuf[3][2];
  __hip_bfloat16* hhbuf[3][2];
  for (int l = 0; l < 3; ++l)
    for (int s = 0; s < 2; ++s) {
      hbuf[l][s]  = (__hip_bfloat16*)carve((size_t)BATCH * HID * 2);
      hhbuf[l][s] = (__hip_bfloat16*)carve((size_t)BATCH * HID * 2);
    }
  float* hsh = (float*)carve((size_t)3 * BATCH * HID * 4);  // fp32 carry shadow -> hT
  __hip_bfloat16* t1b = (__hip_bfloat16*)carve((size_t)BATCH * HID * 2);
  __hip_bfloat16* t2b = (__hip_bfloat16*)carve((size_t)BATCH * HID * 2);
  __hip_bfloat16* xob = (__hip_bfloat16*)carve((size_t)BATCH * XSV * 2);

  // ---- one-time (per call) fp32 -> bf16 conversions
  auto cvt = [&](const float* src, __hip_bfloat16* dst, size_t n) {
    prnn_f32_to_bf16<<<(unsigned)((n + 255) / 256), 256, 0, stream>>>(src, dst, (int)n);
  };
  for (int l = 0; l < 3; ++l) {
    cvt(Wih[l], wihb[l], (size_t)3 * HID * KW);
    cvt(Whh[l], whhb[l], (size_t)3 * HID * KW);
  }
  cvt(W1, w1b, (size_t)HID * KW);
  cvt(W2, w2b, (size_t)HID * KW);
  cvt(W3, w3b, (size_t)XSV * KW);
  cvt(x, xb, (size_t)T_LEN * BATCH * XSV);

  // h0 = zeros
  int cur[3] = {0, 0, 0};
  for (int l = 0; l < 3; ++l)
    prnn_zero_bf16<<<(BATCH * HID + 255) / 256, 256, 0, stream>>>(hbuf[l][0], BATCH * HID);

  const dim3 gruGrid(BATCH / 128, HID / 16);   // 64 blocks x 4 waves
  const dim3 gemGrid(BATCH / 128, 512 / 16);

  auto readout = [&](const __hip_bfloat16* hin, int t, int rr) {
    prnn_gemm_bias<<<gemGrid, 128, RO_LDS, stream>>>(hin, w1b, b1, t1b, (float*)nullptr, 0, 1);
    prnn_gemm_bias<<<gemGrid, 128, RO_LDS, stream>>>(t1b, w2b, b2, t2b, (float*)nullptr, 0, 1);
    float* dst = out + (size_t)t * BATCH * RDEPTH * XSV + (size_t)rr * XSV;  // preds[t, :, rr, :]
    prnn_gemm_bias<<<gemGrid, 128, RO_LDS, stream>>>(t2b, w3b, b3, xob, dst, RDEPTH * XSV, 0);
  };

  for (int t = 0; t < T_LEN; ++t) {
    // carry rnncell on real input x_t (keeps fp32 shadow for hT)
    const __hip_bfloat16* zin = xb + (size_t)t * BATCH * XSV;
    for (int l = 0; l < 3; ++l) {
      prnn_gru_cell<<<gruGrid, 128, GRU_LDS, stream>>>(zin, hbuf[l][cur[l]], wihb[l], whhb[l],
                                                       bih[l], bhh[l], hbuf[l][cur[l] ^ 1],
                                                       hsh + (size_t)l * BATCH * HID);
      cur[l] ^= 1;
      zin = hbuf[l][cur[l]];
    }
    readout(zin, t, 0);

    // future rollout: feed predictions back in (hh starts from carry h)
    int hhcur[3] = {0, 0, 0};
    for (int rr = 1; rr < RDEPTH; ++rr) {
      const __hip_bfloat16* z2 = xob;
      for (int l = 0; l < 3; ++l) {
        const __hip_bfloat16* hin = (rr == 1) ? hbuf[l][cur[l]] : hhbuf[l][hhcur[l]];
        const int oi = (rr == 1) ? 0 : (hhcur[l] ^ 1);
        prnn_gru_cell<<<gruGrid, 128, GRU_LDS, stream>>>(z2, hin, wihb[l], whhb[l],
                                                         bih[l], bhh[l], hhbuf[l][oi],
                                                         (float*)nullptr);
        hhcur[l] = oi;
        z2 = hhbuf[l][oi];
      }
      readout(z2, t, rr);
    }
  }

  // hT = fp32 carry shadow -> tail of d_out
  prnn_copy_f32<<<(3 * BATCH * HID + 255) / 256, 256, 0, stream>>>(
      hsh, out + (size_t)T_LEN * BATCH * RDEPTH * XSV, 3 * BATCH * HID);
}
